// LNN_64381559767881
// MI455X (gfx1250) — matrix operand order repaired and checked by reading, not compiled, not run on hardware
//
#include <hip/hip_runtime.h>
#include <hip/hip_bf16.h>
#include <math.h>

// ---------------------------------------------------------------------------
// Lagrangian NN double-pendulum step, fused into one kernel.
//
// Analytic part (per sample, f32 VALU):
//   c = cos(t1-t2), s = sin(t1-t2)
//   M = [[2,c],[c,1]],  C = [w2*s*(w2-w1), w1*s*(w2-w1)]
//   dT/dq = [-w1*w2*s, +w1*w2*s]
// MLP part (compute bound -> WMMA f16, f32 accumulate):
//   h1 = tanh(feat@W1+b1); h2 = tanh(h1@W2+b2)
//   d2 = W3 * (1-h2^2);    d1 = (d2@W2^T) * (1-h1^2);  dV/dfeat = d1@W1^T
//
// W2 is staged in LDS twice (both orientations row-major along the
// contraction index) so every WMMA B-fragment is two contiguous
// ds_load_b128s; the 136-halfword row stride makes the 16-lane access
// pattern bank-conflict-free (bank step 4 per lane, 4 banks per b128).
// ---------------------------------------------------------------------------

typedef __attribute__((ext_vector_type(16))) _Float16 v16h;
typedef __attribute__((ext_vector_type(8)))  _Float16 v8h;
typedef __attribute__((ext_vector_type(8)))  float    v8f;

#define HID   128
#define SROW  136          // padded f16 row stride (16B aligned, bank-skewed)
#define TPW   4            // 16-sample tiles per wave
#define WAVES 4            // waves per block
#define SPB   (WAVES * TPW * 16)   // samples per block = 256

#if __has_builtin(__builtin_amdgcn_tanhf)
__device__ __forceinline__ float fast_tanh(float x) { return __builtin_amdgcn_tanhf(x); }
#else
__device__ __forceinline__ float fast_tanh(float x) { return tanhf(x); }
#endif

union V16 { v16h v; v8h h[2]; _Float16 e[16]; };

__global__ void __launch_bounds__(WAVES * 32)
lnn_kernel(const float* __restrict__ X,
           const float* __restrict__ W1, const float* __restrict__ b1,
           const float* __restrict__ W2, const float* __restrict__ b2,
           const float* __restrict__ W3, const float* __restrict__ b3,
           float* __restrict__ out, int B)
{
    // W2 in both orientations, contraction index contiguous:
    //   sW2F[n*SROW + k] = W2[k][n]   (forward  B-frag: column n of W2)
    //   sW2B[n*SROW + k] = W2[n][k]   (backward B-frag: row    n of W2)
    __shared__ _Float16 sW2F[HID * SROW];          // 34816 B
    __shared__ _Float16 sW2B[HID * SROW];          // 34816 B
    __shared__ _Float16 sH1[WAVES][16 * SROW];     // 17408 B
    __shared__ _Float16 sD2[WAVES][16 * SROW];     // 17408 B
    __shared__ float    sW1s[4 * HID];             //  2048 B
    __shared__ float    sB1s[HID], sB2s[HID], sW3s[HID];
    __shared__ float    sDV[WAVES][16 * 4];        //  1024 B

    const int tid = threadIdx.x;
    const int nthr = WAVES * 32;

    // ---- stage weights (f32 -> f16 for W2, both orientations) ----
    for (int i = tid; i < HID * HID; i += nthr) {
        const int k = i >> 7, n = i & (HID - 1);
        const _Float16 w = (_Float16)W2[i];
        sW2B[k * SROW + n] = w;   // row-major W2
        sW2F[n * SROW + k] = w;   // W2 transposed
    }
    for (int i = tid; i < 4 * HID; i += nthr) sW1s[i] = W1[i];
    for (int i = tid; i < HID; i += nthr) {
        sB1s[i] = b1[i]; sB2s[i] = b2[i]; sW3s[i] = W3[i];
    }
    __syncthreads();
    (void)b3; // V's constant bias never reaches the gradients

    const int wave  = tid >> 5;
    const int lane  = tid & 31;
    const int ls    = lane & 15;   // sample-in-tile / matrix column
    const int khalf = lane >> 4;   // which K half this lane carries
    const int cbase = khalf * 64;  // layer-1 column range for this lane

    for (int t = 0; t < TPW; ++t) {
        const int tile = (blockIdx.x * WAVES + wave) * TPW + t;
        const int s0   = tile * 16;
        if (s0 >= B) break;

        // prefetch next tile's X rows (global_prefetch_b8)
        if (s0 + 16 + ls < B)
            __builtin_prefetch(&X[(size_t)(s0 + 16 + ls) * 4], 0, 1);

        // ---- per-sample scalars ----
        const int s  = s0 + ls;
        const int sc = (s < B) ? s : (B - 1);
        float4 x = reinterpret_cast<const float4*>(X)[sc];
        const float t1 = x.x, t2 = x.y, w1 = x.z, w2 = x.w;
        const float f0 = sinf(t1), f1 = cosf(t1);
        const float f2 = sinf(t2), f3 = cosf(t2);

        // ---- layer 1 (K=4, VALU) -> h1 f16 tile in LDS ----
        _Float16* h1row = &sH1[wave][ls * SROW];
#pragma unroll
        for (int c8 = 0; c8 < 8; ++c8) {
            v8h hv;
#pragma unroll
            for (int j = 0; j < 8; ++j) {
                const int col = cbase + c8 * 8 + j;
                float a = sB1s[col]
                        + f0 * sW1s[col]           + f1 * sW1s[HID + col]
                        + f2 * sW1s[2 * HID + col] + f3 * sW1s[3 * HID + col];
                hv[j] = (_Float16)fast_tanh(a);
            }
            *(v8h*)&h1row[cbase + c8 * 8] = hv;
        }

        // ---- forward: a2 = h1 @ W2  (8 N-tiles x 4 K-chunks of WMMA) ----
#pragma unroll
        for (int nt = 0; nt < 8; ++nt) {
            v8f acc = {};
            const int n = nt * 16 + ls;
#pragma unroll
            for (int kc = 0; kc < 4; ++kc) {
                V16 a, bm;
                // A frag: row ls of h1, two contiguous 8-f16 runs per lane half
                a.h[0] = *(const v8h*)&sH1[wave][ls * SROW + kc * 32 + khalf * 8];
                a.h[1] = *(const v8h*)&sH1[wave][ls * SROW + kc * 32 + 16 + khalf * 8];
                // B frag: column n of W2 = contiguous run of sW2F
                const v8h* q = (const v8h*)&sW2F[n * SROW + kc * 32 + khalf * 8];
                bm.h[0] = q[0];
                bm.h[1] = q[2];
                acc = __builtin_amdgcn_wmma_f32_16x16x32_f16(
                        false, a.v, false, bm.v, (short)0, acc, false, false);
            }
            // D layout: VGPR r -> row M = r + 8*khalf, col N = n
            const float b2n = sB2s[n], w3n = sW3s[n];
#pragma unroll
            for (int r = 0; r < 8; ++r) {
                const int m  = r + 8 * khalf;
                const float h2 = fast_tanh(acc[r] + b2n);
                sD2[wave][m * SROW + n] = (_Float16)(w3n * (1.0f - h2 * h2));
            }
        }

        // ---- backward: g1 = d2 @ W2^T ; fold into dV/dfeat accumulators ----
        float dv[8][4];
#pragma unroll
        for (int r = 0; r < 8; ++r)
#pragma unroll
            for (int k = 0; k < 4; ++k) dv[r][k] = 0.0f;

#pragma unroll
        for (int nt = 0; nt < 8; ++nt) {
            v8f acc = {};
            const int n = nt * 16 + ls;   // output index i of g1
#pragma unroll
            for (int kc = 0; kc < 4; ++kc) {
                V16 a, bm;
                a.h[0] = *(const v8h*)&sD2[wave][ls * SROW + kc * 32 + khalf * 8];
                a.h[1] = *(const v8h*)&sD2[wave][ls * SROW + kc * 32 + 16 + khalf * 8];
                // B^T frag: row n of W2 = contiguous run of sW2B
                const v8h* q = (const v8h*)&sW2B[n * SROW + kc * 32 + khalf * 8];
                bm.h[0] = q[0];
                bm.h[1] = q[2];
                acc = __builtin_amdgcn_wmma_f32_16x16x32_f16(
                        false, a.v, false, bm.v, (short)0, acc, false, false);
            }
            const float w1n0 = sW1s[n],           w1n1 = sW1s[HID + n];
            const float w1n2 = sW1s[2 * HID + n], w1n3 = sW1s[3 * HID + n];
#pragma unroll
            for (int r = 0; r < 8; ++r) {
                const int m   = r + 8 * khalf;
                const float h1v = (float)sH1[wave][m * SROW + n];
                const float d1  = acc[r] * (1.0f - h1v * h1v);
                dv[r][0] += w1n0 * d1;  dv[r][1] += w1n1 * d1;
                dv[r][2] += w1n2 * d1;  dv[r][3] += w1n3 * d1;
            }
        }

        // ---- reduce dv across the 16 lanes of each half ----
#pragma unroll
        for (int off = 1; off < 16; off <<= 1)
#pragma unroll
            for (int r = 0; r < 8; ++r)
#pragma unroll
                for (int k = 0; k < 4; ++k)
                    dv[r][k] += __shfl_xor(dv[r][k], off, 32);

        if (ls == 0) {
#pragma unroll
            for (int r = 0; r < 8; ++r)
#pragma unroll
                for (int k = 0; k < 4; ++k)
                    sDV[wave][(khalf * 8 + r) * 4 + k] = dv[r][k];
        }

        // ---- closed-form Euler-Lagrange solve + store ----
        const float dvdf0 = sDV[wave][ls * 4 + 0];
        const float dvdf1 = sDV[wave][ls * 4 + 1];
        const float dvdf2 = sDV[wave][ls * 4 + 2];
        const float dvdf3 = sDV[wave][ls * 4 + 3];
        const float dVdt1 = dvdf0 * f1 - dvdf1 * f0;
        const float dVdt2 = dvdf2 * f3 - dvdf3 * f2;
        const float s12 = f0 * f3 - f1 * f2;   // sin(t1-t2)
        const float c12 = f1 * f3 + f0 * f2;   // cos(t1-t2)
        const float dLdt1 = -w1 * w2 * s12 - dVdt1;
        const float dLdt2 =  w1 * w2 * s12 - dVdt2;
        const float r1 = dLdt1 - w2 * s12 * (w2 - w1);
        const float r2 = dLdt2 - w1 * s12 * (w2 - w1);
        const float inv_det = 1.0f / (2.0f - c12 * c12);
        const float qdd1 = (r1 - c12 * r2) * inv_det;
        const float qdd2 = (2.0f * r2 - c12 * r1) * inv_det;

        if (khalf == 0 && s < B) {
            float4 o; o.x = w1; o.y = w2; o.z = qdd1; o.w = qdd2;
            reinterpret_cast<float4*>(out)[s] = o;
        }
    }
}

extern "C" void kernel_launch(void* const* d_in, const int* in_sizes, int n_in,
                              void* d_out, int out_size, void* d_ws, size_t ws_size,
                              hipStream_t stream)
{
    (void)n_in; (void)out_size; (void)d_ws; (void)ws_size;
    const float* X  = (const float*)d_in[0];
    const float* W1 = (const float*)d_in[1];
    const float* b1 = (const float*)d_in[2];
    const float* W2 = (const float*)d_in[3];
    const float* b2 = (const float*)d_in[4];
    const float* W3 = (const float*)d_in[5];
    const float* b3 = (const float*)d_in[6];
    float* out = (float*)d_out;

    const int B = in_sizes[0] / 4;
    const int blocks = (B + SPB - 1) / SPB;
    lnn_kernel<<<blocks, WAVES * 32, 0, stream>>>(X, W1, b1, W2, b2, W3, b3, out, B);
}
